// SkipConnect_24154896073559
// MI455X (gfx1250) — matrix operand order repaired and checked by reading, not compiled, not run on hardware
//
#include <hip/hip_runtime.h>
#include <hip/hip_bf16.h>
#include <stdint.h>

// ---------------------------------------------------------------------------
// CDNA5 (gfx1250) wave32 WMMA implementation.
// Shapes: B=8, C=128, c=64, H=W=64, n=HW=4096.
// Matrix work on v_wmma_f32_16x16x32_bf16; flash-attention K/V tiles streamed
// with global_load_async_to_lds_b128 + ASYNCcnt double buffering.
// ---------------------------------------------------------------------------

typedef __attribute__((ext_vector_type(16))) __bf16 v16bf;
typedef __attribute__((ext_vector_type(8)))  float  v8f;

#define WMMA_BF16(A, B, C) \
  __builtin_amdgcn_wmma_f32_16x16x32_bf16(false, (A), false, (B), (short)0, (C), false, false)

#define NB 8
#define NHW 4096
#define HC 64      // half channels
#define FC 128     // full channels

// ---- async global -> LDS copy (CDNA5, tracked by ASYNCcnt) ----------------
// GV addressing: 64-bit global address in a VGPR pair, LDS byte offset in a
// VGPR (low 32 bits of the generic shared pointer = wave-relative LDS offset,
// ISA section 10.2 aperture mapping). 16 bytes per lane.
static __device__ __forceinline__ void async_copy_b128(void* lds_ptr, const void* gaddr) {
  uint32_t ldsoff = (uint32_t)(uintptr_t)lds_ptr;
  uint64_t ga = (uint64_t)(uintptr_t)gaddr;
  asm volatile("global_load_async_to_lds_b128 %0, %1, off"
               :: "v"(ldsoff), "v"(ga)
               : "memory");
}
#define ASYNC_WAIT_4() asm volatile("s_wait_asynccnt 0x4" ::: "memory")
#define ASYNC_WAIT_0() asm volatile("s_wait_asynccnt 0x0" ::: "memory")

// ---- WMMA fragment loaders (ISA 7.12.2 16-bit layouts, wave32) ------------
// A matrix 16x32 (MxK): lanes 0-15 = M rows, lane half selects K sub-block.
//   VGPR v<4  : K = 2v  + 8*half  (pairs K,K+1)
//   VGPR v>=4 : K = 2v+8 + 8*half
// Source is row-major (row = m, k contiguous), row stride = ld elements.
static __device__ __forceinline__ v16bf load_a_frag(const __bf16* src, int ld) {
  int lane = threadIdx.x & 31;
  int m    = lane & 15;
  int half = lane >> 4;
  v16bf a;
#pragma unroll
  for (int v = 0; v < 8; ++v) {
    int k = ((v < 4) ? (2 * v) : (2 * v + 8)) + half * 8;
    a[2 * v]     = src[m * ld + k];
    a[2 * v + 1] = src[m * ld + k + 1];
  }
  return a;
}

// B matrix 32x16 (KxN): column n=lane%16, same K order as A:
//   VGPR v<4 : K = 2v + 8*half; VGPR v>=4 : K = 2v+8 + 8*half (pairs K,K+1).
// Source is stored TRANSPOSED: srcT[col][k], col stride = ldt elements,
// so the per-lane K pairs are contiguous in memory.
static __device__ __forceinline__ v16bf load_b_frag_t(const __bf16* srcT, int ldt) {
  int lane = threadIdx.x & 31;
  int n    = lane & 15;
  int half = lane >> 4;
  v16bf b;
#pragma unroll
  for (int v = 0; v < 8; ++v) {
    int k = ((v < 4) ? (2 * v) : (2 * v + 8)) + half * 8;
    b[2 * v]     = srcT[n * ldt + k];
    b[2 * v + 1] = srcT[n * ldt + k + 1];
  }
  return b;
}

// ---------------------------------------------------------------------------
// Kernel 1: prep. Per (b, n) position (128 threads):
//  - shortcuts (fp32, NHWC per branch)
//  - KV projections kv = W * s + bias for both branches
//  - bf16 repacks: Q1[n][c], K1[n][c], V1T[c][n], U2T[c][n], K2T[c][n], V2[n][c]
// ---------------------------------------------------------------------------
__global__ __launch_bounds__(128) void sc_prep_kernel(
    const float* __restrict__ xup, const float* __restrict__ xskip,
    const float* __restrict__ kv1w, const float* __restrict__ kv1b,
    const float* __restrict__ kv2w, const float* __restrict__ kv2b,
    __bf16* __restrict__ Q1, __bf16* __restrict__ K1, __bf16* __restrict__ V1T,
    __bf16* __restrict__ U2T, __bf16* __restrict__ K2T, __bf16* __restrict__ V2,
    float* __restrict__ SC1, float* __restrict__ SC2) {
  __shared__ float s1[HC];
  __shared__ float s2[HC];
  int pos = blockIdx.x;
  int b = pos >> 12;
  int n = pos & (NHW - 1);
  int tid = threadIdx.x;
  size_t bc = (size_t)b * FC * NHW;
  if (tid < HC) {
    float u1 = xup[bc + (size_t)tid * NHW + n];
    float k1 = xskip[bc + (size_t)tid * NHW + n];
    float u2 = xup[bc + (size_t)(tid + HC) * NHW + n];
    float k2 = xskip[bc + (size_t)(tid + HC) * NHW + n];
    s1[tid] = k1;
    s2[tid] = k2;
    size_t nh = ((size_t)b * NHW + n) * HC + tid;
    Q1[nh] = (__bf16)u1;
    U2T[((size_t)b * HC + tid) * NHW + n] = (__bf16)u2;
    SC1[nh] = u1 + k1;
    SC2[nh] = u2 + k2;
  }
  __syncthreads();
  float d1 = kv1b[tid];
  float d2 = kv2b[tid];
#pragma unroll 8
  for (int c = 0; c < HC; ++c) {
    d1 += kv1w[tid * HC + c] * s1[c];
    d2 += kv2w[tid * HC + c] * s2[c];
  }
  if (tid < HC) {
    K1[((size_t)b * NHW + n) * HC + tid] = (__bf16)d1;
    K2T[((size_t)b * HC + tid) * NHW + n] = (__bf16)d2;
  } else {
    int j = tid - HC;
    V1T[((size_t)b * HC + j) * NHW + n] = (__bf16)d1;
    V2[((size_t)b * NHW + n) * HC + j] = (__bf16)d2;
  }
}

// ---------------------------------------------------------------------------
// Kernel 2: flash attention (branch 1). Grid (32 q-tiles, 8 b), 256 thr.
// Each of 8 waves owns 16 query rows; 64-key K/V tiles double-buffered in LDS
// via global_load_async_to_lds_b128 (4 async b128 per thread per tile,
// prefetch distance 2, s_wait_asynccnt gating).
// Writes sout + shortcut1 into XC (NHWC bf16, channels 0..63).
// ---------------------------------------------------------------------------
__global__ __launch_bounds__(256) void flash1_kernel(
    const __bf16* __restrict__ Q1, const __bf16* __restrict__ K1,
    const __bf16* __restrict__ V1T, const float* __restrict__ SC1,
    __bf16* __restrict__ XC) {
  __shared__ __bf16 Klds[2][64 * 64];       // [buf][key][ch]
  __shared__ __bf16 Vlds[2][64 * 64];       // [buf][ch][key] (transposed)
  __shared__ __bf16 Plds[8 * 16 * 64];      // per-wave P staging [row][key]
  int b = blockIdx.y;
  int q0 = blockIdx.x * 128;
  int tid = threadIdx.x;
  int wave = tid >> 5;
  int lane = tid & 31;
  int qr = q0 + wave * 16;

  const char* Kb = (const char*)(K1 + (size_t)b * NHW * HC);
  const char* Vb = (const char*)(V1T + (size_t)b * HC * NHW);

  const __bf16* qbase = Q1 + ((size_t)b * NHW + qr) * HC;
  v16bf qa0 = load_a_frag(qbase, HC);
  v16bf qa1 = load_a_frag(qbase + 32, HC);

  v8f acc[4];
  float rm[8], rl[8];
#pragma unroll
  for (int t = 0; t < 4; ++t)
#pragma unroll
    for (int r = 0; r < 8; ++r) acc[t][r] = 0.f;
#pragma unroll
  for (int r = 0; r < 8; ++r) { rm[r] = -1e30f; rl[r] = 0.f; }

  // Issue one 64-key tile (K: 8KB, V: 8KB) -> 4 async b128 per thread.
  auto issue_tile = [&](int j0, int buf) {
    // K tile [key][ch]: 512 uint4 vectors, vec = tid and tid+256.
    char* lk = (char*)&Klds[buf][0] + tid * 16;
    const char* gk = Kb + (size_t)j0 * 128 + tid * 16;
    async_copy_b128(lk, gk);
    async_copy_b128(lk + 4096, gk + 4096);
    // V tile [ch][key]: vec i -> ch = i>>3, part = i&7; second vec ch+32.
    char* lv = (char*)&Vlds[buf][0] + tid * 16;
    const char* gv = Vb + (size_t)(tid >> 3) * (NHW * 2) + (size_t)j0 * 2 + (size_t)(tid & 7) * 16;
    async_copy_b128(lv, gv);
    async_copy_b128(lv + 4096, gv + (size_t)32 * (NHW * 2));
  };

  issue_tile(0, 0);
  issue_tile(64, 1);

  for (int jt = 0; jt < 64; ++jt) {
    int buf = jt & 1;
    // Async loads complete in order per wave: <=4 outstanding means the
    // current tile's 4 transfers are done (the 4 left are the next tile's).
    if (jt < 63) { ASYNC_WAIT_4(); } else { ASYNC_WAIT_0(); }
    __syncthreads();   // all waves' tiles resident

    v8f S[4];
#pragma unroll
    for (int nt = 0; nt < 4; ++nt) {
      v8f s;
#pragma unroll
      for (int r = 0; r < 8; ++r) s[r] = 0.f;
      s = WMMA_BF16(qa0, load_b_frag_t(&Klds[buf][nt * 16 * 64], 64), s);
      s = WMMA_BF16(qa1, load_b_frag_t(&Klds[buf][nt * 16 * 64 + 32], 64), s);
      S[nt] = s;
    }

    // Online softmax: row r lives in VGPR r across a 16-lane half.
#pragma unroll
    for (int r = 0; r < 8; ++r) {
      float mx = fmaxf(fmaxf(S[0][r], S[1][r]), fmaxf(S[2][r], S[3][r]));
      mx = fmaxf(mx, __shfl_xor(mx, 1));
      mx = fmaxf(mx, __shfl_xor(mx, 2));
      mx = fmaxf(mx, __shfl_xor(mx, 4));
      mx = fmaxf(mx, __shfl_xor(mx, 8));
      float nm = fmaxf(rm[r], mx);
      float scale = __expf(rm[r] - nm);
      rm[r] = nm;
      float rs = 0.f;
#pragma unroll
      for (int nt = 0; nt < 4; ++nt) {
        float p = __expf(S[nt][r] - nm);
        S[nt][r] = p;
        rs += p;
      }
      rs += __shfl_xor(rs, 1);
      rs += __shfl_xor(rs, 2);
      rs += __shfl_xor(rs, 4);
      rs += __shfl_xor(rs, 8);
      rl[r] = rl[r] * scale + rs;
#pragma unroll
      for (int nt = 0; nt < 4; ++nt) acc[nt][r] *= scale;
    }

    // Stage P (C-layout -> row-major) in per-wave LDS; per-wave DS ops are
    // in-order, so the following A-frag loads see the stores.
    __bf16* pw = Plds + wave * 16 * 64;
#pragma unroll
    for (int r = 0; r < 8; ++r) {
      int m = r + ((lane >> 4) << 3);
#pragma unroll
      for (int nt = 0; nt < 4; ++nt)
        pw[m * 64 + nt * 16 + (lane & 15)] = (__bf16)S[nt][r];
    }
    v16bf pa0 = load_a_frag(pw, 64);
    v16bf pa1 = load_a_frag(pw + 32, 64);
#pragma unroll
    for (int nt = 0; nt < 4; ++nt) {
      acc[nt] = WMMA_BF16(pa0, load_b_frag_t(&Vlds[buf][nt * 16 * 64], 64), acc[nt]);
      acc[nt] = WMMA_BF16(pa1, load_b_frag_t(&Vlds[buf][nt * 16 * 64 + 32], 64), acc[nt]);
    }

    __syncthreads();   // all waves done reading `buf` before it is refilled
    if (jt < 62) issue_tile((jt + 2) * 64, buf);
  }

  // Epilogue: O/l + shortcut1 -> XC (NHWC bf16, ch 0..63)
#pragma unroll
  for (int r = 0; r < 8; ++r) {
    int m = r + ((lane >> 4) << 3);
    int q = qr + m;
    float inv = 1.0f / rl[r];
#pragma unroll
    for (int nt = 0; nt < 4; ++nt) {
      int ch = nt * 16 + (lane & 15);
      size_t nh = ((size_t)b * NHW + q) * HC + ch;
      float v = acc[nt][r] * inv + SC1[nh];
      XC[((size_t)b * NHW + q) * FC + ch] = (__bf16)v;
    }
  }
}

// ---------------------------------------------------------------------------
// Kernel 3: channel-attention Gram matrix (branch 2a). One WG per batch.
// G = U2^T K2 (64x64, K=4096) via WMMA, softmax over rows, store A^T bf16.
// ---------------------------------------------------------------------------
__global__ __launch_bounds__(256) void gram2_kernel(
    const __bf16* __restrict__ U2T, const __bf16* __restrict__ K2T,
    __bf16* __restrict__ AT) {
  __shared__ __bf16 Ulds[64 * 128];   // [c][n-chunk]
  __shared__ __bf16 Klds[64 * 128];   // [d][n-chunk]
  __shared__ float Gs[64 * 64];
  int b = blockIdx.x;
  int tid = threadIdx.x;
  int wave = tid >> 5;
  int lane = tid & 31;
  int t0 = wave * 2;                  // two 16x16 tiles per wave, same c-row
  int tc = t0 >> 2, td0 = t0 & 3, td1 = (t0 + 1) & 3;

  v8f acc0, acc1;
#pragma unroll
  for (int r = 0; r < 8; ++r) { acc0[r] = 0.f; acc1[r] = 0.f; }

  for (int n0 = 0; n0 < NHW; n0 += 128) {
    __syncthreads();
    for (int i = tid; i < 1024; i += 256) {
      int row = i >> 4, part = i & 15;
      ((uint4*)Ulds)[i] = ((const uint4*)(U2T + ((size_t)b * HC + row) * NHW + n0))[part];
      ((uint4*)Klds)[i] = ((const uint4*)(K2T + ((size_t)b * HC + row) * NHW + n0))[part];
    }
    __syncthreads();
#pragma unroll
    for (int kk = 0; kk < 4; ++kk) {
      v16bf a = load_a_frag(Ulds + tc * 16 * 128 + kk * 32, 128);
      acc0 = WMMA_BF16(a, load_b_frag_t(Klds + td0 * 16 * 128 + kk * 32, 128), acc0);
      acc1 = WMMA_BF16(a, load_b_frag_t(Klds + td1 * 16 * 128 + kk * 32, 128), acc1);
    }
  }
  __syncthreads();
#pragma unroll
  for (int r = 0; r < 8; ++r) {
    int m = r + ((lane >> 4) << 3);
    Gs[(tc * 16 + m) * 64 + td0 * 16 + (lane & 15)] = acc0[r];
    Gs[(tc * 16 + m) * 64 + td1 * 16 + (lane & 15)] = acc1[r];
  }
  __syncthreads();
  if (tid < 64) {
    int d = tid;
    float mx = -1e30f;
    for (int c = 0; c < 64; ++c) mx = fmaxf(mx, Gs[c * 64 + d]);
    float s = 0.f;
    for (int c = 0; c < 64; ++c) s += __expf(Gs[c * 64 + d] - mx);
    float inv = 1.0f / s;
    for (int c = 0; c < 64; ++c)
      AT[((size_t)b * 64 + d) * 64 + c] = (__bf16)(__expf(Gs[c * 64 + d] - mx) * inv);
  }
}

// ---------------------------------------------------------------------------
// Kernel 4: apply channel attention (branch 2b): cout = V2 @ A + shortcut2.
// Writes XC channels 64..127.
// ---------------------------------------------------------------------------
__global__ __launch_bounds__(256) void cout2_kernel(
    const __bf16* __restrict__ V2, const __bf16* __restrict__ AT,
    const float* __restrict__ SC2, __bf16* __restrict__ XC) {
  int b = blockIdx.y;
  int q0 = blockIdx.x * 128;
  int tid = threadIdx.x;
  int wave = tid >> 5;
  int lane = tid & 31;
  int qr = q0 + wave * 16;
  const __bf16* vbase = V2 + ((size_t)b * NHW + qr) * HC;
  const __bf16* abase = AT + (size_t)b * 64 * 64;
  v16bf a0 = load_a_frag(vbase, HC);
  v16bf a1 = load_a_frag(vbase + 32, HC);
#pragma unroll
  for (int dt = 0; dt < 4; ++dt) {
    v8f acc;
#pragma unroll
    for (int r = 0; r < 8; ++r) acc[r] = 0.f;
    acc = WMMA_BF16(a0, load_b_frag_t(abase + dt * 16 * 64, 64), acc);
    acc = WMMA_BF16(a1, load_b_frag_t(abase + dt * 16 * 64 + 32, 64), acc);
#pragma unroll
    for (int r = 0; r < 8; ++r) {
      int m = r + ((lane >> 4) << 3);
      int q = qr + m;
      int ch = dt * 16 + (lane & 15);
      size_t nh = ((size_t)b * NHW + q) * HC + ch;
      float v = acc[r] + SC2[nh];
      XC[((size_t)b * NHW + q) * FC + HC + ch] = (__bf16)v;
    }
  }
}

// ---------------------------------------------------------------------------
// Kernel 5: conv weight repack (OIHW f32 -> [tap][o][i] bf16) + BN fold.
// ---------------------------------------------------------------------------
__global__ __launch_bounds__(256) void convprep_kernel(
    const float* __restrict__ convw, const float* __restrict__ gamma,
    const float* __restrict__ beta, const float* __restrict__ mean,
    const float* __restrict__ var, __bf16* __restrict__ WP,
    float* __restrict__ scale, float* __restrict__ bias) {
  int i = blockIdx.x * 256 + threadIdx.x;
  if (i < 9 * FC * FC) {
    int t = i / (FC * FC);
    int rem = i - t * (FC * FC);
    int o = rem >> 7;
    int ii = rem & 127;
    WP[(size_t)(t * FC + o) * FC + ii] = (__bf16)convw[(size_t)(o * FC + ii) * 9 + t];
  }
  if (i < FC) {
    float inv = rsqrtf(var[i] + 1e-5f);
    float sc = inv * gamma[i];
    scale[i] = sc;
    bias[i] = beta[i] - mean[i] * sc;
  }
}

// ---------------------------------------------------------------------------
// Kernel 6: 3x3 conv as implicit GEMM + BN + ReLU. Grid (64 y, 8 b), 256 thr.
// LDS holds 3 input rows x 66 pixels x 128 ch (NHWC bf16, zero-padded halo).
// Wave w computes output channels [16w,16w+16) for 4 x-tiles of 16 pixels.
// ---------------------------------------------------------------------------
__global__ __launch_bounds__(256) void conv_bn_relu_kernel(
    const __bf16* __restrict__ XC, const __bf16* __restrict__ WP,
    const float* __restrict__ scale, const float* __restrict__ bias,
    float* __restrict__ out) {
  __shared__ __bf16 Xlds[3 * 66 * FC];   // 50688 bytes
  int b = blockIdx.y;
  int y = blockIdx.x;
  int tid = threadIdx.x;
  int wave = tid >> 5;
  int lane = tid & 31;

  for (int i = tid; i < 3 * 66 * 16; i += 256) {   // 16 uint4 per pixel
    int pix = i >> 4, part = i & 15;
    int ry = pix / 66, xx = pix - ry * 66;
    int yy = y + ry - 1, x = xx - 1;
    uint4 v = make_uint4(0u, 0u, 0u, 0u);
    if (yy >= 0 && yy < 64 && x >= 0 && x < 64)
      v = ((const uint4*)(XC + ((size_t)b * NHW + yy * 64 + x) * FC))[part];
    ((uint4*)Xlds)[i] = v;
  }
  __syncthreads();

  int o0 = wave * 16;
  v8f acc[4];
#pragma unroll
  for (int t = 0; t < 4; ++t)
#pragma unroll
    for (int r = 0; r < 8; ++r) acc[t][r] = 0.f;

  for (int t = 0; t < 9; ++t) {
    int ky = t / 3, kx = t - ky * 3;
#pragma unroll
    for (int kk = 0; kk < 4; ++kk) {
      v16bf a = load_a_frag(WP + (size_t)(t * FC + o0) * FC + kk * 32, FC);
#pragma unroll
      for (int xt = 0; xt < 4; ++xt) {
        // input pixel index in LDS: xx = x_out + kx (halo offset +1 folded in)
        const __bf16* bsrc = Xlds + (size_t)(ky * 66 + xt * 16 + kx) * FC + kk * 32;
        acc[xt] = WMMA_BF16(a, load_b_frag_t(bsrc, FC), acc[xt]);
      }
    }
  }

#pragma unroll
  for (int r = 0; r < 8; ++r) {
    int m = r + ((lane >> 4) << 3);
    int o = o0 + m;
    float sc = scale[o], bi = bias[o];
#pragma unroll
    for (int xt = 0; xt < 4; ++xt) {
      int x = xt * 16 + (lane & 15);
      float v = acc[xt][r] * sc + bi;
      out[(((size_t)b * FC + o) * 64 + y) * 64 + x] = fmaxf(v, 0.f);
    }
  }
}

// ---------------------------------------------------------------------------
// Host-side launch
// ---------------------------------------------------------------------------
extern "C" void kernel_launch(void* const* d_in, const int* in_sizes, int n_in,
                              void* d_out, int out_size, void* d_ws, size_t ws_size,
                              hipStream_t stream) {
  const float* xup   = (const float*)d_in[0];
  const float* xskip = (const float*)d_in[1];
  const float* kv1w  = (const float*)d_in[2];
  const float* kv1b  = (const float*)d_in[3];
  const float* kv2w  = (const float*)d_in[4];
  const float* kv2b  = (const float*)d_in[5];
  const float* convw = (const float*)d_in[6];
  const float* gamma = (const float*)d_in[7];
  const float* beta  = (const float*)d_in[8];
  const float* mean  = (const float*)d_in[9];
  const float* var   = (const float*)d_in[10];
  float* out = (float*)d_out;

  char* ws = (char*)d_ws;
  size_t off = 0;
  auto carve = [&](size_t bytes) -> char* {
    char* p = ws + off;
    off += (bytes + 255) & ~(size_t)255;
    return p;
  };
  const size_t nhHalfF32  = (size_t)NB * NHW * HC * sizeof(float);     // 8 MB
  const size_t nhHalfBF16 = (size_t)NB * NHW * HC * sizeof(__bf16);    // 4 MB

  float*  SC1 = (float*)carve(nhHalfF32);
  float*  SC2 = (float*)carve(nhHalfF32);
  __bf16* Q1  = (__bf16*)carve(nhHalfBF16);
  __bf16* K1  = (__bf16*)carve(nhHalfBF16);
  __bf16* V1T = (__bf16*)carve(nhHalfBF16);
  __bf16* U2T = (__bf16*)carve(nhHalfBF16);
  __bf16* K2T = (__bf16*)carve(nhHalfBF16);
  __bf16* V2  = (__bf16*)carve(nhHalfBF16);
  __bf16* AT  = (__bf16*)carve((size_t)NB * 64 * 64 * sizeof(__bf16));
  __bf16* XC  = (__bf16*)carve((size_t)NB * NHW * FC * sizeof(__bf16));
  __bf16* WP  = (__bf16*)carve((size_t)9 * FC * FC * sizeof(__bf16));
  float*  SCL = (float*)carve(FC * sizeof(float));
  float*  BIA = (float*)carve(FC * sizeof(float));

  sc_prep_kernel<<<NB * NHW, 128, 0, stream>>>(
      xup, xskip, kv1w, kv1b, kv2w, kv2b, Q1, K1, V1T, U2T, K2T, V2, SC1, SC2);

  convprep_kernel<<<(9 * FC * FC + 255) / 256, 256, 0, stream>>>(
      convw, gamma, beta, mean, var, WP, SCL, BIA);

  flash1_kernel<<<dim3(32, NB), 256, 0, stream>>>(Q1, K1, V1T, SC1, XC);

  gram2_kernel<<<NB, 256, 0, stream>>>(U2T, K2T, AT);

  cout2_kernel<<<dim3(32, NB), 256, 0, stream>>>(V2, AT, SC2, XC);

  conv_bn_relu_kernel<<<dim3(64, NB), 256, 0, stream>>>(XC, WP, SCL, BIA, out);
}
